// NeuroLiteTokenizer_54812372632230
// MI455X (gfx1250) — compile-verified, matrix operand order checked by reading, not run on hardware
//
#include <hip/hip_runtime.h>
#include <hip/hip_bf16.h>

// ---------------------------------------------------------------------------
// NeuroLite tokenizer for MI455X (gfx1250, wave32, WMMA).
// - All matmuls (projections, codebook distance scans, contrastive sim matrix)
//   run through v_wmma_f32_16x16x32_f16.
// - All GEMM operands are pre-converted to f16; weights are pre-transposed to
//   [N,K] so every mode uses one n-major B layout => tiles are row-contiguous
//   64B strips in both global and LDS, enabling GLOBAL_LOAD_ASYNC_TO_LDS_B128
//   double-buffered staging (ASYNCcnt-tracked) when the builtin is available.
// - Argmin / exp-rowsum fused into the WMMA epilogue: the [B,V] distance and
//   [B,B] similarity matrices never exist in memory.
// ---------------------------------------------------------------------------

#define Bdim 8192
#define Hdim 1024
#define SEMV 8192
#define DETV 8192
#define RVQV 1024
#define NQr  4

typedef __attribute__((ext_vector_type(16))) _Float16 v16h;
typedef __attribute__((ext_vector_type(8)))  float    v8f;

#if defined(__has_builtin)
#if __has_builtin(__builtin_amdgcn_global_load_async_to_lds_b128) && \
    __has_builtin(__builtin_amdgcn_s_wait_asynccnt)
#define USE_ASYNC 1
#endif
#endif
#ifndef USE_ASYNC
#define USE_ASYNC 0
#endif

#if USE_ASYNC
// The builtin takes generic pointers to 128-bit int vectors (per hipcc
// diagnostic: "int __attribute__((__vector_size__(4 * sizeof(int)))) *").
typedef __attribute__((__vector_size__(16))) int b128v;

// Stage one 64x32-half tile pair (A and B) via async global->LDS b128 copies.
// 64 rows x 64B per tile = 256 x 16B chunks; 128 threads -> 2 chunks each.
// 4 async instructions per wave per call (ASYNCcnt += 4).
static __device__ __forceinline__ void
issue_tile(const _Float16* gA, int lda, const _Float16* gB, int ldb,
           _Float16* lA, _Float16* lB, int tid) {
#pragma unroll
  for (int t = 0; t < 2; ++t) {
    int c = tid + t * 128;
    int r = c >> 2, j = (c & 3) * 8;              // j in halfs (16B chunks)
    __builtin_amdgcn_global_load_async_to_lds_b128(
        (b128v*)(gA + (size_t)r * lda + j), (b128v*)(lA + r * 32 + j), 0, 0);
    __builtin_amdgcn_global_load_async_to_lds_b128(
        (b128v*)(gB + (size_t)r * ldb + j), (b128v*)(lB + r * 32 + j), 0, 0);
  }
}
#endif

// Load a 16x32 f16 fragment (A-layout per CDNA5 ISA 7.12.2) from an LDS tile
// stored row-major with ld=32 halfs (lowers to packed ds_load_b128).
static __device__ __forceinline__ v16h frag_ld(const _Float16* base, int row, int lane) {
  union { v16h v; unsigned u[8]; } f;
  const unsigned* p = (const unsigned*)(base + row * 32);
  const int hi4 = (lane >> 4) << 2;               // lanes 16..31 take K+8 halves
#pragma unroll
  for (int vg = 0; vg < 8; ++vg)
    f.u[vg] = p[((vg & 4) << 1) + hi4 + (vg & 3)];
  return f.v;
}

static __device__ __forceinline__ void
wmma_step(const _Float16* As, const _Float16* Bs, int lane, int wm, int wn, v8f acc[2][2]) {
  v16h a0 = frag_ld(As, wm + (lane & 15), lane);
  v16h a1 = frag_ld(As, wm + 16 + (lane & 15), lane);
  v16h b0 = frag_ld(Bs, wn + (lane & 15), lane);
  v16h b1 = frag_ld(Bs, wn + 16 + (lane & 15), lane);
  acc[0][0] = __builtin_amdgcn_wmma_f32_16x16x32_f16(false, a0, false, b0, (short)0, acc[0][0], false, false);
  acc[0][1] = __builtin_amdgcn_wmma_f32_16x16x32_f16(false, a0, false, b1, (short)0, acc[0][1], false, false);
  acc[1][0] = __builtin_amdgcn_wmma_f32_16x16x32_f16(false, a1, false, b0, (short)0, acc[1][0], false, false);
  acc[1][1] = __builtin_amdgcn_wmma_f32_16x16x32_f16(false, a1, false, b1, (short)0, acc[1][1], false, false);
}

// MODE 0: C[M,N] = A@B^T + bscale*bias  (B given as [N,K]); optional f16 copy.
// MODE 1: per-row argmin over n of (cnorm[n] - 2*A·B[n]) -> packed atomicMin.
// MODE 2: negsum[m] += sum_n exp(A·B[n]*invtemp); diag contributes exp(0)=1.
template <int MODE>
__global__ void __launch_bounds__(128)
gemm_wmma(const _Float16* __restrict__ A, int lda,
          const _Float16* __restrict__ Bn,
          const float* __restrict__ bias, float bscale,
          float* __restrict__ C, _Float16* __restrict__ Ch, int ldch,
          const float* __restrict__ cnorm, unsigned long long* __restrict__ minkey,
          int N, int K, float invtemp) {
  __shared__ __align__(16) _Float16 As[2][64 * 32];
  __shared__ __align__(16) _Float16 Bs[2][64 * 32];
  const int tid = threadIdx.x, lane = tid & 31, wave = tid >> 5;
  const int wm = (wave >> 1) * 32, wn = (wave & 1) * 32;
  const int m0 = blockIdx.y * 64, n0 = blockIdx.x * 64;
  const _Float16* gA = A + (size_t)m0 * lda;
  const _Float16* gB = Bn + (size_t)n0 * K;
  const int nIter = K >> 5;

  v8f acc[2][2] = {};

#if USE_ASYNC
  issue_tile(gA, lda, gB, K, As[0], Bs[0], tid);
  for (int it = 0; it + 1 < nIter; ++it) {
    issue_tile(gA + (it + 1) * 32, lda, gB + (it + 1) * 32, K,
               As[(it + 1) & 1], Bs[(it + 1) & 1], tid);
    __builtin_amdgcn_s_wait_asynccnt(4);          // oldest batch (current tile) done
    __syncthreads();
    wmma_step(As[it & 1], Bs[it & 1], lane, wm, wn, acc);
    __syncthreads();                              // reads done before buffer reuse
  }
  __builtin_amdgcn_s_wait_asynccnt(0);
  __syncthreads();
  wmma_step(As[(nIter - 1) & 1], Bs[(nIter - 1) & 1], lane, wm, wn, acc);
#else
  for (int it = 0; it < nIter; ++it) {
    __syncthreads();
#pragma unroll 4
    for (int i = tid; i < 2048; i += 128) {
      int r = i >> 5, c = i & 31;
      As[0][r * 32 + c] = gA[(size_t)r * lda + it * 32 + c];
      Bs[0][r * 32 + c] = gB[(size_t)r * K + it * 32 + c];
    }
    if (it + 1 < nIter)                           // global_prefetch_b8
      __builtin_prefetch(gA + (size_t)(tid >> 1) * lda + (it + 1) * 32, 0, 3);
    __syncthreads();
    wmma_step(As[0], Bs[0], lane, wm, wn, acc);
  }
#endif

  const int ln15 = lane & 15;
  const int hi8  = (lane >> 4) << 3;
#pragma unroll
  for (int tm = 0; tm < 2; ++tm) {
#pragma unroll
    for (int tn = 0; tn < 2; ++tn) {
#pragma unroll
      for (int r = 0; r < 8; ++r) {
        int gm = m0 + wm + tm * 16 + r + hi8;     // C/D layout: VGPR r, lane halves
        int gn = n0 + wn + tn * 16 + ln15;
        float val = acc[tm][tn][r];
        if (MODE == 0) {
          float y = val + (bias ? bias[gn] * bscale : 0.0f);
          if (C)  C[(size_t)gm * N + gn] = y;
          if (Ch) Ch[(size_t)gm * ldch + gn] = (_Float16)y;
        } else if (MODE == 1) {
          float s = cnorm[gn] - 2.0f * val;
          int   mi = gn;
#pragma unroll
          for (int off = 8; off; off >>= 1) {     // min across 16 cols
            float os = __shfl_xor(s, off, 32);
            int   oi = __shfl_xor(mi, off, 32);
            if (os < s || (os == s && oi < mi)) { s = os; mi = oi; }
          }
          if (ln15 == 0) {
            unsigned us = __float_as_uint(s);
            us = (us & 0x80000000u) ? ~us : (us | 0x80000000u);  // order-preserving
            unsigned long long key = ((unsigned long long)us << 32) | (unsigned)mi;
            atomicMin(&minkey[gm], key);
          }
        } else {
          float e = (gm == gn) ? 1.0f : __expf(val * invtemp);
#pragma unroll
          for (int off = 8; off; off >>= 1) e += __shfl_xor(e, off, 32);
          if (ln15 == 0) atomicAdd(&C[gm], e);
        }
      }
    }
  }
}

static __device__ __forceinline__ float block_reduce(float v, float* sh) {
  int tid = threadIdx.x;
  sh[tid] = v; __syncthreads();
  for (int s = blockDim.x >> 1; s > 0; s >>= 1) {
    if (tid < s) sh[tid] += sh[tid + s];
    __syncthreads();
  }
  float r = sh[0]; __syncthreads();
  return r;
}

__global__ void __launch_bounds__(256)
f2h_kernel(const float* __restrict__ s, _Float16* __restrict__ d) {
  size_t i = (size_t)blockIdx.x * 256 + threadIdx.x;
  d[i] = (_Float16)s[i];
}

// W [K,N] row-major -> WT [N,K] f16 (so GEMM B-side is always n-major)
__global__ void __launch_bounds__(256)
f2hT_kernel(const float* __restrict__ s, _Float16* __restrict__ d, int rows, int cols) {
  size_t i = (size_t)blockIdx.x * 256 + threadIdx.x;
  int r = (int)(i / cols), c = (int)(i % cols);
  d[(size_t)c * rows + r] = (_Float16)s[i];
}

__global__ void __launch_bounds__(256)
add_kernel(float* __restrict__ o, _Float16* __restrict__ oh,
           const float* __restrict__ a, const float* __restrict__ b) {
  size_t i = (size_t)blockIdx.x * 256 + threadIdx.x;
  float v = a[i] + b[i];
  o[i] = v; oh[i] = (_Float16)v;
}

// combined = LN(0.5*(s+u)); emitted as f16 (only a GEMM consumes it)
__global__ void __launch_bounds__(256)
combine_ln_kernel(const float* __restrict__ s, const float* __restrict__ u,
                  const float* __restrict__ g, const float* __restrict__ b,
                  _Float16* __restrict__ outh) {
  __shared__ float sh[256];
  size_t row = blockIdx.x;
  float x[4]; float sum = 0.f, sq = 0.f;
#pragma unroll
  for (int j = 0; j < 4; ++j) {
    int k = threadIdx.x + j * 256;
    x[j] = 0.5f * (s[row * Hdim + k] + u[row * Hdim + k]);
    sum += x[j]; sq += x[j] * x[j];
  }
  sum = block_reduce(sum, sh); sq = block_reduce(sq, sh);
  float mu = sum * (1.0f / Hdim);
  float inv = rsqrtf(sq * (1.0f / Hdim) - mu * mu + 1e-5f);
#pragma unroll
  for (int j = 0; j < 4; ++j) {
    int k = threadIdx.x + j * 256;
    outh[row * Hdim + k] = (_Float16)((x[j] - mu) * inv * g[k] + b[k]);
  }
}

template <bool GELU>
__global__ void __launch_bounds__(256)
ln_kernel(const float* __restrict__ x, float* __restrict__ yf, _Float16* __restrict__ yh,
          const float* __restrict__ g, const float* __restrict__ b) {
  __shared__ float sh[256];
  size_t row = blockIdx.x;
  float v[4]; float sum = 0.f, sq = 0.f;
#pragma unroll
  for (int j = 0; j < 4; ++j) {
    int k = threadIdx.x + j * 256;
    v[j] = x[row * Hdim + k];
    sum += v[j]; sq += v[j] * v[j];
  }
  sum = block_reduce(sum, sh); sq = block_reduce(sq, sh);
  float mu = sum * (1.0f / Hdim);
  float inv = rsqrtf(sq * (1.0f / Hdim) - mu * mu + 1e-5f);
#pragma unroll
  for (int j = 0; j < 4; ++j) {
    int k = threadIdx.x + j * 256;
    float y = (v[j] - mu) * inv * g[k] + b[k];
    if (GELU)  // tanh approximation (jax.nn.gelu default)
      y = 0.5f * y * (1.0f + tanhf(0.7978845608f * (y + 0.044715f * y * y * y)));
    if (yf) yf[row * Hdim + k] = y;
    yh[row * Hdim + k] = (_Float16)y;
  }
}

__global__ void __launch_bounds__(256)
cnorm_kernel(const float* __restrict__ cb, float* __restrict__ cn) {
  __shared__ float sh[256];
  size_t row = blockIdx.x;
  float s = 0.f;
  for (int k = threadIdx.x; k < Hdim; k += 256) {
    float v = cb[row * Hdim + k]; s += v * v;
  }
  s = block_reduce(s, sh);
  if (threadIdx.x == 0) cn[row] = s;
}

// Straight-through VQ forward: q = cb[idx]; writes f16 q into concat slot,
// or accumulates quant / updates residual (f32 + f16); commitment loss via
// block reduce + atomics.
__global__ void __launch_bounds__(256)
vq_apply(const float* __restrict__ z, const float* __restrict__ cb,
         const unsigned long long* __restrict__ minkey,
         _Float16* __restrict__ qh, long qh_stride, long qh_off,
         float* __restrict__ quant_add, _Float16* __restrict__ quanth,
         float* __restrict__ res_out, _Float16* __restrict__ resh,
         float* __restrict__ idx_out, float cost, float* __restrict__ loss_acc) {
  __shared__ float sh[256];
  size_t gid = (size_t)blockIdx.x * 256 + threadIdx.x;
  int m = (int)(gid >> 10), k = (int)(gid & 1023);
  int idx = (int)(unsigned)(minkey[m] & 0xFFFFFFFFull);
  float zz = z[gid];
  float q = cb[(size_t)idx * Hdim + k];
  if (qh) qh[(size_t)m * qh_stride + qh_off + k] = (_Float16)q;
  if (quant_add) {
    float nv = quant_add[gid] + q;
    quant_add[gid] = nv;
    if (quanth) quanth[gid] = (_Float16)nv;
  }
  float d = zz - q;
  if (res_out) res_out[gid] = d;
  if (resh)    resh[gid] = (_Float16)d;
  if (idx_out && k == 0) idx_out[m] = (float)idx;
  float s = block_reduce(d * d, sh);
  if (threadIdx.x == 0)
    atomicAdd(loss_acc, cost * s * (1.0f / ((float)Bdim * (float)Hdim)));
}

// Row-normalize both modalities (f16 out, feeds sim GEMM) + positive cosine
__global__ void __launch_bounds__(256)
norm_kernel(const float* __restrict__ a, const float* __restrict__ b,
            _Float16* __restrict__ na, _Float16* __restrict__ nb,
            float* __restrict__ possum) {
  __shared__ float sh[256];
  size_t row = blockIdx.x;
  float xa[4], xb[4]; float sa = 0.f, sb = 0.f, sab = 0.f;
#pragma unroll
  for (int j = 0; j < 4; ++j) {
    int k = threadIdx.x + j * 256;
    xa[j] = a[row * Hdim + k]; xb[j] = b[row * Hdim + k];
    sa += xa[j] * xa[j]; sb += xb[j] * xb[j]; sab += xa[j] * xb[j];
  }
  sa = block_reduce(sa, sh); sb = block_reduce(sb, sh); sab = block_reduce(sab, sh);
  float ia = rsqrtf(sa), ib = rsqrtf(sb);
#pragma unroll
  for (int j = 0; j < 4; ++j) {
    int k = threadIdx.x + j * 256;
    na[row * Hdim + k] = (_Float16)(xa[j] * ia);
    nb[row * Hdim + k] = (_Float16)(xb[j] * ib);
  }
  if (threadIdx.x == 0) possum[row] = sab * ia * ib;
}

__global__ void __launch_bounds__(256)
align_final(const float* __restrict__ possum, const float* __restrict__ negsum,
            float* __restrict__ acc1) {
  __shared__ float sh[256];
  int i = blockIdx.x * 256 + threadIdx.x;
  float p = __expf(possum[i] * 10.0f);            // 1/TEMP
  float v = logf(p / (p + negsum[i] + 1e-8f));
  float s = block_reduce(v, sh);
  if (threadIdx.x == 0) atomicAdd(acc1, -s * (1.0f / (float)Bdim));
}

__global__ void writeback(const float* __restrict__ acc, float* __restrict__ out) {
  out[0] = acc[0];   // vq_loss
  out[1] = acc[1];   // align_loss
}

extern "C" void kernel_launch(void* const* d_in, const int* in_sizes, int n_in,
                              void* d_out_v, int out_size, void* d_ws, size_t ws_size,
                              hipStream_t stream) {
  (void)in_sizes; (void)n_in; (void)out_size; (void)ws_size;
  const float* text  = (const float*)d_in[0];
  const float* image = (const float*)d_in[1];
  const float* Wv    = (const float*)d_in[6];
  const float* bv    = (const float*)d_in[7];
  const float* Wo    = (const float*)d_in[8];
  const float* bo    = (const float*)d_in[9];
  const float* ln_g  = (const float*)d_in[10];
  const float* ln_b  = (const float*)d_in[11];
  const float* postW = (const float*)d_in[12];
  const float* postb = (const float*)d_in[13];
  const float* semcb = (const float*)d_in[14];
  const float* detcb = (const float*)d_in[15];
  const float* W1    = (const float*)d_in[16];
  const float* b1    = (const float*)d_in[17];
  const float* ln1g  = (const float*)d_in[18];
  const float* ln1b  = (const float*)d_in[19];
  const float* W2    = (const float*)d_in[20];
  const float* b2    = (const float*)d_in[21];
  const float* ln2g  = (const float*)d_in[22];
  const float* ln2b  = (const float*)d_in[23];
  const float* rvqcb = (const float*)d_in[24];
  const float* Wout  = (const float*)d_in[25];
  const float* bout  = (const float*)d_in[26];

  float* out        = (float*)d_out_v;
  float* out_tokens = out;
  float* out_semidx = out + (size_t)Bdim * Hdim;
  float* out_detidx = out_semidx + Bdim;
  float* out_residx = out_detidx + Bdim;
  float* out_losses = out_residx + (size_t)NQr * Bdim;

  const size_t BH = (size_t)Bdim * Hdim;
  const size_t HH = (size_t)Hdim * Hdim;
  // f32 scratch
  float* buf0 = (float*)d_ws;                     // s -> aligned -> refined/res
  float* buf1 = buf0 + BH;                        // u -> h -> h2 -> quant
  // f16 scratch
  _Float16* ah0 = (_Float16*)(buf1 + BH);         // 2*BH halfs: s/combined/concat/quant/nb
  _Float16* ah1 = ah0 + 2 * BH;                   // BH halfs: v/aligned/h/refined-res/na
  _Float16* wvT   = ah1 + BH;
  _Float16* woT   = wvT + HH;
  _Float16* postT = woT + HH;
  _Float16* w1T   = postT + HH;                   // 2*HH
  _Float16* w2T   = w1T + 2 * HH;
  _Float16* woutT = w2T + HH;
  _Float16* semh  = woutT + HH;                   // SEMV*Hdim
  _Float16* deth  = semh + (size_t)SEMV * Hdim;
  _Float16* rvqh  = deth + (size_t)DETV * Hdim;
  float* cnorm = (float*)(rvqh + (size_t)RVQV * Hdim);
  unsigned long long* minkey = (unsigned long long*)(cnorm + 8192);
  float* possum = (float*)(minkey + Bdim);
  float* negsum = possum + Bdim;
  float* acc    = negsum + Bdim;                  // [0]=vq_loss, [1]=align_loss

  dim3 blk(128);
  dim3 gN(Hdim / 64, Bdim / 64);
  dim3 gSem(SEMV / 64, Bdim / 64);
  dim3 gRvq(RVQV / 64, Bdim / 64);
  dim3 gSim(Bdim / 64, Bdim / 64);
  int ew_grid = (int)(BH / 256);

  (void)hipMemsetAsync(acc, 0, 2 * sizeof(float), stream);
  (void)hipMemsetAsync(negsum, 0, Bdim * sizeof(float), stream);

  // --- one-time f16 conversions (weights transposed to [N,K])
  f2hT_kernel<<<(int)(HH / 256), 256, 0, stream>>>(Wv, wvT, Hdim, Hdim);
  f2hT_kernel<<<(int)(HH / 256), 256, 0, stream>>>(Wo, woT, Hdim, Hdim);
  f2hT_kernel<<<(int)(HH / 256), 256, 0, stream>>>(postW, postT, Hdim, Hdim);
  f2hT_kernel<<<(int)(2 * HH / 256), 256, 0, stream>>>(W1, w1T, 2 * Hdim, Hdim);
  f2hT_kernel<<<(int)(HH / 256), 256, 0, stream>>>(W2, w2T, Hdim, Hdim);
  f2hT_kernel<<<(int)(HH / 256), 256, 0, stream>>>(Wout, woutT, Hdim, Hdim);
  f2h_kernel<<<(int)((size_t)SEMV * Hdim / 256), 256, 0, stream>>>(semcb, semh);
  f2h_kernel<<<(int)((size_t)DETV * Hdim / 256), 256, 0, stream>>>(detcb, deth);
  f2h_kernel<<<(int)((size_t)RVQV * Hdim / 256), 256, 0, stream>>>(rvqcb, rvqh);

  // --- collapsed cross attention: s=t+i; v=s@Wv+2bv; u=v@Wo+2bo
  add_kernel<<<ew_grid, 256, 0, stream>>>(buf0, ah0, text, image);
  gemm_wmma<0><<<gN, blk, 0, stream>>>(ah0, Hdim, wvT, bv, 2.0f, nullptr, ah1, Hdim, nullptr, nullptr, Hdim, Hdim, 0.f);
  gemm_wmma<0><<<gN, blk, 0, stream>>>(ah1, Hdim, woT, bo, 2.0f, buf1, nullptr, 0, nullptr, nullptr, Hdim, Hdim, 0.f);
  combine_ln_kernel<<<Bdim, 256, 0, stream>>>(buf0, buf1, ln_g, ln_b, ah0);
  gemm_wmma<0><<<gN, blk, 0, stream>>>(ah0, Hdim, postT, postb, 1.0f, buf0, ah1, Hdim, nullptr, nullptr, Hdim, Hdim, 0.f);

  // --- semantic VQ (fused distance-argmin GEMM, V=8192); q -> concat slot 0
  cnorm_kernel<<<SEMV, 256, 0, stream>>>(semcb, cnorm);
  (void)hipMemsetAsync(minkey, 0xFF, Bdim * 8, stream);
  gemm_wmma<1><<<gSem, blk, 0, stream>>>(ah1, Hdim, semh, nullptr, 0.f, nullptr, nullptr, 0, cnorm, minkey, SEMV, Hdim, 0.f);
  vq_apply<<<ew_grid, 256, 0, stream>>>(buf0, semcb, minkey, ah0, 2 * Hdim, 0,
                                        nullptr, nullptr, nullptr, nullptr, out_semidx, 0.25f, acc);

  // --- detail VQ (cost = 1.5*CC); q -> concat slot 1
  cnorm_kernel<<<DETV, 256, 0, stream>>>(detcb, cnorm);
  (void)hipMemsetAsync(minkey, 0xFF, Bdim * 8, stream);
  gemm_wmma<1><<<gSem, blk, 0, stream>>>(ah1, Hdim, deth, nullptr, 0.f, nullptr, nullptr, 0, cnorm, minkey, DETV, Hdim, 0.f);
  vq_apply<<<ew_grid, 256, 0, stream>>>(buf0, detcb, minkey, ah0, 2 * Hdim, Hdim,
                                        nullptr, nullptr, nullptr, nullptr, out_detidx, 0.375f, acc);

  // --- refinement MLP on the materialized f16 concat (lda = 2H)
  gemm_wmma<0><<<gN, blk, 0, stream>>>(ah0, 2 * Hdim, w1T, b1, 1.0f, buf1, nullptr, 0, nullptr, nullptr, Hdim, 2 * Hdim, 0.f);
  ln_kernel<true><<<Bdim, 256, 0, stream>>>(buf1, nullptr, ah1, ln1g, ln1b);
  gemm_wmma<0><<<gN, blk, 0, stream>>>(ah1, Hdim, w2T, b2, 1.0f, buf1, nullptr, 0, nullptr, nullptr, Hdim, Hdim, 0.f);
  ln_kernel<false><<<Bdim, 256, 0, stream>>>(buf1, buf0, ah1, ln2g, ln2b);  // refined = initial residual

  // --- residual VQ chain (shared codebook, 4 stages)
  (void)hipMemsetAsync(buf1, 0, BH * sizeof(float), stream);      // quant accumulator
  cnorm_kernel<<<RVQV, 256, 0, stream>>>(rvqcb, cnorm);
  for (int st = 0; st < NQr; ++st) {
    (void)hipMemsetAsync(minkey, 0xFF, Bdim * 8, stream);
    gemm_wmma<1><<<gRvq, blk, 0, stream>>>(ah1, Hdim, rvqh, nullptr, 0.f, nullptr, nullptr, 0, cnorm, minkey, RVQV, Hdim, 0.f);
    vq_apply<<<ew_grid, 256, 0, stream>>>(buf0, rvqcb, minkey, nullptr, 0, 0,
                                          buf1, (st == NQr - 1) ? ah0 : nullptr,
                                          buf0, ah1, out_residx + (size_t)st * Bdim, 0.25f, acc);
  }

  // --- output projection straight into d_out (A = f16 quant)
  gemm_wmma<0><<<gN, blk, 0, stream>>>(ah0, Hdim, woutT, bout, 1.0f, out_tokens, nullptr, 0, nullptr, nullptr, Hdim, Hdim, 0.f);

  // --- contrastive alignment loss (fused exp/rowsum GEMM over [B,B])
  norm_kernel<<<Bdim, 256, 0, stream>>>(text, image, ah1, ah0 + BH, possum);
  gemm_wmma<2><<<gSim, blk, 0, stream>>>(ah1, Hdim, ah0 + BH, nullptr, 0.f, negsum, nullptr, 0, nullptr, nullptr, Bdim, Hdim, 10.0f);
  align_final<<<Bdim / 256, 256, 0, stream>>>(possum, negsum, acc + 1);
  writeback<<<1, 1, 0, stream>>>(acc, out_losses);
}
